// RoPEAttnBlock_11733850653068
// MI455X (gfx1250) — matrix operand order repaired and checked by reading, not compiled, not run on hardware
//
#include <hip/hip_runtime.h>
#include <hip/hip_bf16.h>
#include <math.h>

#define SEQ 4096
#define DM  1024
#define NH  16
#define DH  64
#define H3  3072
#define HFF 4096

typedef unsigned short u16;
typedef unsigned int   u32;
typedef __attribute__((ext_vector_type(16))) __bf16 v16bf;
typedef __attribute__((ext_vector_type(8)))  float  v8f;

// fp32 -> bf16 round-to-nearest-even
static __device__ __forceinline__ u16 f2bf(float f) {
  union { float f; u32 u; } v; v.f = f;
  u32 r = v.u + 0x7FFFu + ((v.u >> 16) & 1u);
  return (u16)(r >> 16);
}

union FragU { v16bf v; uint4 q[2]; };

// Build a 16-element bf16 fragment from two 16-byte chunks.
static __device__ __forceinline__ v16bf load_frag(const u16* p0, const u16* p1) {
  FragU f;
  f.q[0] = *(const uint4*)p0;
  f.q[1] = *(const uint4*)p1;
  return f.v;
}

static __device__ __forceinline__ v8f wmma_bf16(v16bf a, v16bf b, v8f c) {
  return __builtin_amdgcn_wmma_f32_16x16x32_bf16(false, a, false, b, (short)0, c, false, false);
}

// ---------------------------------------------------------------------------
// Weight fp32 [K][N] -> bf16 pre-swizzled B-fragment layout:
// tile (nt,kt): element (lane l, halfword h) = W[kt*32 + 16*(l/16) + h][nt*16 + l%16]
// stored contiguously so GEMM loads 32B/lane with two b128s.
// ---------------------------------------------------------------------------
__global__ __launch_bounds__(256) void k_wswz(const float* __restrict__ W,
                                              u16* __restrict__ out, int K, int N) {
  int idx = blockIdx.x * 256 + threadIdx.x;
  if (idx >= K * N) return;
  int tile = idx >> 9;
  int rem  = idx & 511;
  int l = rem >> 4;
  int h = rem & 15;
  int ktiles = K >> 5;
  int nt = tile / ktiles;
  int kt = tile - nt * ktiles;
  int n = nt * 16 + (l & 15);
  int k = kt * 32 + (l >> 4) * 16 + h;
  out[idx] = f2bf(W[(size_t)k * N + n]);
}

// ---------------------------------------------------------------------------
// LayerNorm: one 256-thread block per row of 1024, writes bf16.
// ---------------------------------------------------------------------------
__global__ __launch_bounds__(256) void k_ln(const float* __restrict__ X,
                                            const float* __restrict__ G,
                                            const float* __restrict__ Bp,
                                            u16* __restrict__ O) {
  __shared__ float r1[256], r2[256];
  int tid = threadIdx.x;
  size_t base = (size_t)blockIdx.x * DM;
  float v[4]; float s = 0.f, ss = 0.f;
#pragma unroll
  for (int i = 0; i < 4; ++i) { v[i] = X[base + tid + 256 * i]; s += v[i]; ss += v[i] * v[i]; }
  r1[tid] = s; r2[tid] = ss; __syncthreads();
  for (int off = 128; off > 0; off >>= 1) {
    if (tid < off) { r1[tid] += r1[tid + off]; r2[tid] += r2[tid + off]; }
    __syncthreads();
  }
  float mean = r1[0] * (1.f / DM);
  float var  = r2[0] * (1.f / DM) - mean * mean;
  float rstd = rsqrtf(var + 1e-5f);
#pragma unroll
  for (int i = 0; i < 4; ++i) {
    int c = tid + 256 * i;
    O[base + c] = f2bf((v[i] - mean) * rstd * G[c] + Bp[c]);
  }
}

// ---------------------------------------------------------------------------
// Fragment loads for one K-step of the GEMM (A row-major bf16, B pre-swizzled)
// ---------------------------------------------------------------------------
static __device__ __forceinline__ void gemm_load(const u16* __restrict__ A,
                                                 const u16* __restrict__ Bs,
                                                 int row0, int lm, int g, int lane,
                                                 int ni, int ktiles, int K, int kt,
                                                 v16bf a[2], v16bf b[4]) {
#pragma unroll
  for (int t = 0; t < 2; ++t) {
    const u16* p = A + (size_t)(row0 + t * 16 + lm) * K + kt * 32 + 16 * g;
    a[t] = load_frag(p, p + 8);
  }
#pragma unroll
  for (int j = 0; j < 4; ++j) {
    const u16* p = Bs + ((size_t)((ni * 4 + j) * ktiles + kt) << 9) + lane * 16;
    b[j] = load_frag(p, p + 8);
  }
}

// ---------------------------------------------------------------------------
// bf16 WMMA GEMM: C[M,N] = A[M,K] (row-major bf16) x Bs (pre-swizzled bf16).
// Wave tile 32x64 (2x4 tiles of 16x16, 8 WMMA per 32-deep K step), with
// double-buffered fragment prefetch so loads overlap WMMA execution.
// MODE 0: store f32               (QKV)
// MODE 1: +residual, f32          (Wo)
// MODE 2: +bias, exact GELU, bf16 (FFN up)
// MODE 3: +bias +residual, f32    (FFN down -> output)
// ---------------------------------------------------------------------------
template<int MODE>
__global__ __launch_bounds__(256) void k_gemm(const u16* __restrict__ A,
                                              const u16* __restrict__ Bs,
                                              float* __restrict__ outF,
                                              u16* __restrict__ outB,
                                              const float* __restrict__ bias,
                                              const float* __restrict__ resid,
                                              int M, int N, int K) {
  int tid = threadIdx.x;
  int wave = tid >> 5, lane = tid & 31, g = lane >> 4, lm = lane & 15;
  int task = blockIdx.x * 8 + wave;
  int ntn = N >> 6;
  if (task >= (M >> 5) * ntn) return;   // wave-uniform
  int mi = task / ntn, ni = task - mi * ntn;
  int ktiles = K >> 5;
  v8f z = {0.f, 0.f, 0.f, 0.f, 0.f, 0.f, 0.f, 0.f};
  v8f acc[2][4];
#pragma unroll
  for (int t = 0; t < 2; ++t)
#pragma unroll
    for (int j = 0; j < 4; ++j) acc[t][j] = z;
  int row0 = mi * 32;

  // software pipeline: current + next fragment sets
  v16bf aC[2], bC[4], aN[2], bN[4];
  gemm_load(A, Bs, row0, lm, g, lane, ni, ktiles, K, 0, aC, bC);
  for (int kt = 0; kt < ktiles; ++kt) {
    if (kt + 1 < ktiles)
      gemm_load(A, Bs, row0, lm, g, lane, ni, ktiles, K, kt + 1, aN, bN);
#pragma unroll
    for (int t = 0; t < 2; ++t)
#pragma unroll
      for (int j = 0; j < 4; ++j)
        acc[t][j] = wmma_bf16(aC[t], bC[j], acc[t][j]);
#pragma unroll
    for (int t = 0; t < 2; ++t) aC[t] = aN[t];
#pragma unroll
    for (int j = 0; j < 4; ++j) bC[j] = bN[j];
  }

#pragma unroll
  for (int t = 0; t < 2; ++t) {
#pragma unroll
    for (int j = 0; j < 4; ++j) {
      int col = ni * 64 + j * 16 + lm;
#pragma unroll
      for (int r = 0; r < 8; ++r) {
        int row = row0 + t * 16 + r + 8 * g;
        size_t o = (size_t)row * N + col;
        float v = acc[t][j][r];
        if (MODE == 0) {
          outF[o] = v;
        } else if (MODE == 1) {
          outF[o] = v + resid[o];
        } else if (MODE == 2) {
          v += bias[col];
          v = 0.5f * v * (1.f + erff(v * 0.70710678118f));
          outB[o] = f2bf(v);
        } else {
          outF[o] = v + bias[col] + resid[o];
        }
      }
    }
  }
}

// ---------------------------------------------------------------------------
// RoPE: qkv f32 [n][3*DM] -> Q,K bf16 [H][n][dh] (rotated), V bf16 transposed
// [H][dh][n] so PV B-fragments are contiguous per lane.
// ---------------------------------------------------------------------------
__global__ __launch_bounds__(256) void k_rope(const float* __restrict__ qkv,
                                              u16* __restrict__ Qb,
                                              u16* __restrict__ Kb,
                                              u16* __restrict__ Vt) {
  int t  = blockIdx.x * 256 + threadIdx.x;
  int i2 = t & 31;
  int n  = (t >> 5) & (SEQ - 1);
  int h  = t >> 17;
  float fr = __expf(-(float)i2 * 0.28782313662425f); // ln(10000)/32
  float th = (float)n * fr;
  float sn, cs;
  sincosf(th, &sn, &cs);
  size_t base = (size_t)n * H3 + h * DH + 2 * i2;
  float q0 = qkv[base],          q1 = qkv[base + 1];
  float k0 = qkv[base + DM],     k1 = qkv[base + DM + 1];
  float v0 = qkv[base + 2 * DM], v1 = qkv[base + 2 * DM + 1];
  size_t qd = ((size_t)h * SEQ + n) * DH + 2 * i2;
  Qb[qd]     = f2bf(q0 * cs - q1 * sn);
  Qb[qd + 1] = f2bf(q0 * sn + q1 * cs);
  Kb[qd]     = f2bf(k0 * cs - k1 * sn);
  Kb[qd + 1] = f2bf(k0 * sn + k1 * cs);
  size_t vd = ((size_t)h * DH + 2 * i2) * SEQ + n;
  Vt[vd]       = f2bf(v0);
  Vt[vd + SEQ] = f2bf(v1);
}

// ---------------------------------------------------------------------------
// Causal flash attention: one wave per (head, 16-row q tile).
// Per 32-key chunk: ALL 8 fragments (4x K^T + 4x V) are loaded up front so
// the V latency hides behind S-WMMAs + softmax; then 4 WMMA for S = QK^T,
// online softmax staged through per-wave LDS, 4 WMMA for O += P V.
// LDS ops are in-order within a wave -> no barriers needed.
// ---------------------------------------------------------------------------
__global__ __launch_bounds__(256) void k_attn(const u16* __restrict__ Qb,
                                              const u16* __restrict__ Kb,
                                              const u16* __restrict__ Vt,
                                              u16* __restrict__ Ob) {
  __shared__ alignas(16) float Sb[8][16][32];
  __shared__ alignas(16) u16   Pb[8][16][32];
  __shared__ alignas(16) float Mst[8][16];
  __shared__ alignas(16) float Lst[8][16];
  __shared__ alignas(16) float Ast[8][16];
  int tid = threadIdx.x;
  int w = tid >> 5, lane = tid & 31, g = lane >> 4, lm = lane & 15;
  int task = blockIdx.x * 8 + w;
  int head = task >> 8;
  int qt = task & 255;
  const u16* Qh = Qb + (size_t)head * SEQ * DH;
  const u16* Kh = Kb + (size_t)head * SEQ * DH;
  const u16* Vh = Vt + (size_t)head * DH * SEQ;

  v16bf qa[2];
#pragma unroll
  for (int kt = 0; kt < 2; ++kt) {
    const u16* p = Qh + (size_t)(qt * 16 + lm) * DH + kt * 32 + 16 * g;
    qa[kt] = load_frag(p, p + 8);
  }
  if (lane < 16) { Mst[w][lane] = -INFINITY; Lst[w][lane] = 0.f; }
  __asm__ volatile("" ::: "memory");

  v8f z = {0.f, 0.f, 0.f, 0.f, 0.f, 0.f, 0.f, 0.f};
  v8f oacc[4] = {z, z, z, z};
  int nch = (qt >> 1) + 1;
  for (int c = 0; c < nch; ++c) {
    // ---- batch all global loads for this chunk (16 x b128 in flight) ----
    v16bf kf[2][2];   // [s sub-tile][dh chunk]
    v16bf vf[4];      // [dh 16-col tile]
#pragma unroll
    for (int s = 0; s < 2; ++s)
#pragma unroll
      for (int kt = 0; kt < 2; ++kt) {
        const u16* p = Kh + (size_t)(c * 32 + s * 16 + lm) * DH + kt * 32 + 16 * g;
        kf[s][kt] = load_frag(p, p + 8);
      }
#pragma unroll
    for (int nt = 0; nt < 4; ++nt) {
      const u16* pv = Vh + (size_t)(nt * 16 + lm) * SEQ + c * 32 + 16 * g;
      vf[nt] = load_frag(pv, pv + 8);
    }

    // ---- S = Q K^T, scaled + causal-masked, staged to LDS ----
    v8f sacc[2];
#pragma unroll
    for (int s = 0; s < 2; ++s) {
      sacc[s] = wmma_bf16(qa[0], kf[s][0], z);
      sacc[s] = wmma_bf16(qa[1], kf[s][1], sacc[s]);
    }
#pragma unroll
    for (int s = 0; s < 2; ++s) {
#pragma unroll
      for (int r = 0; r < 8; ++r) {
        int row = qt * 16 + r + 8 * g;
        int col = c * 32 + s * 16 + lm;
        float vv = sacc[s][r] * 0.125f;  // 1/sqrt(64)
        if (col > row) vv = -INFINITY;
        Sb[w][r + 8 * g][s * 16 + lm] = vv;
      }
    }
    __asm__ volatile("" ::: "memory");

    // ---- row-wise online softmax (lanes 0..15 own one row each) ----
    if (lane < 16) {
      const float4* s4 = (const float4*)&Sb[w][lane][0];
      float sv[32];
#pragma unroll
      for (int j = 0; j < 8; ++j) {
        float4 tq = s4[j];
        sv[4 * j] = tq.x; sv[4 * j + 1] = tq.y; sv[4 * j + 2] = tq.z; sv[4 * j + 3] = tq.w;
      }
      float mo = Mst[w][lane];
      float mx = mo;
#pragma unroll
      for (int j = 0; j < 32; ++j) mx = fmaxf(mx, sv[j]);
      float alpha = __expf(mo - mx);
      float rs = 0.f;
      u32* pp = (u32*)&Pb[w][lane][0];
#pragma unroll
      for (int j = 0; j < 16; ++j) {
        float p0 = __expf(sv[2 * j]     - mx);
        float p1 = __expf(sv[2 * j + 1] - mx);
        rs += p0 + p1;
        pp[j] = (u32)f2bf(p0) | ((u32)f2bf(p1) << 16);
      }
      Lst[w][lane] = Lst[w][lane] * alpha + rs;
      Mst[w][lane] = mx;
      Ast[w][lane] = alpha;
    }
    __asm__ volatile("" ::: "memory");

    // ---- P fragment + rescale + O += P V (V already resident) ----
    const u16* pr = &Pb[w][lm][0];
    v16bf pf = load_frag(pr + 16 * g, pr + 16 * g + 8);
    float4 a0 = *(const float4*)&Ast[w][8 * g];
    float4 a1 = *(const float4*)&Ast[w][8 * g + 4];
    float alp[8] = {a0.x, a0.y, a0.z, a0.w, a1.x, a1.y, a1.z, a1.w};
#pragma unroll
    for (int nt = 0; nt < 4; ++nt) {
#pragma unroll
      for (int r = 0; r < 8; ++r) oacc[nt][r] *= alp[r];
      oacc[nt] = wmma_bf16(pf, vf[nt], oacc[nt]);
    }
  }
  __asm__ volatile("" ::: "memory");
  float4 l0 = *(const float4*)&Lst[w][8 * g];
  float4 l1 = *(const float4*)&Lst[w][8 * g + 4];
  float li[8] = {l0.x, l0.y, l0.z, l0.w, l1.x, l1.y, l1.z, l1.w};
#pragma unroll
  for (int r = 0; r < 8; ++r) li[r] = 1.f / li[r];
#pragma unroll
  for (int nt = 0; nt < 4; ++nt) {
    int col = head * DH + nt * 16 + lm;
#pragma unroll
    for (int r = 0; r < 8; ++r) {
      int row = qt * 16 + r + 8 * g;
      Ob[(size_t)row * DM + col] = f2bf(oacc[nt][r] * li[r]);
    }
  }
}

// ---------------------------------------------------------------------------
extern "C" void kernel_launch(void* const* d_in, const int* in_sizes, int n_in,
                              void* d_out, int out_size, void* d_ws, size_t ws_size,
                              hipStream_t stream) {
  (void)in_sizes; (void)n_in; (void)out_size; (void)ws_size;
  const float* x    = (const float*)d_in[0];
  const float* ln1g = (const float*)d_in[1];
  const float* ln1b = (const float*)d_in[2];
  const float* ln2g = (const float*)d_in[3];
  const float* ln2b = (const float*)d_in[4];
  const float* Wqkv = (const float*)d_in[5];
  const float* Wo   = (const float*)d_in[6];
  const float* W1   = (const float*)d_in[7];
  const float* b1   = (const float*)d_in[8];
  const float* W2   = (const float*)d_in[9];
  const float* b2   = (const float*)d_in[10];
  float* out = (float*)d_out;

  char* ws = (char*)d_ws;
  size_t off = 0;
  auto alloc = [&](size_t bytes) {
    char* p = ws + off;
    off = (off + bytes + 255) & ~(size_t)255;
    return p;
  };
  u16*   Wqkv_s = (u16*)alloc((size_t)DM * H3 * 2);
  u16*   Wo_s   = (u16*)alloc((size_t)DM * DM * 2);
  u16*   W1_s   = (u16*)alloc((size_t)DM * HFF * 2);
  u16*   W2_s   = (u16*)alloc((size_t)HFF * DM * 2);
  u16*   hb     = (u16*)alloc((size_t)SEQ * DM * 2);   // LN1 out; reused as LN2 out
  float* qkv    = (float*)alloc((size_t)SEQ * H3 * 4); // reused as f1 (bf16, smaller)
  u16*   f1     = (u16*)qkv;
  u16*   Qb     = (u16*)alloc((size_t)NH * SEQ * DH * 2);
  u16*   Kb     = (u16*)alloc((size_t)NH * SEQ * DH * 2);
  u16*   Vt     = (u16*)alloc((size_t)NH * SEQ * DH * 2);
  u16*   Ob     = (u16*)alloc((size_t)SEQ * DM * 2);
  float* x1     = (float*)alloc((size_t)SEQ * DM * 4);

  // one-time weight bf16 swizzles
  k_wswz<<<(DM * H3) / 256, 256, 0, stream>>>(Wqkv, Wqkv_s, DM, H3);
  k_wswz<<<(DM * DM) / 256, 256, 0, stream>>>(Wo, Wo_s, DM, DM);
  k_wswz<<<(DM * HFF) / 256, 256, 0, stream>>>(W1, W1_s, DM, HFF);
  k_wswz<<<(HFF * DM) / 256, 256, 0, stream>>>(W2, W2_s, HFF, DM);
  // LN1
  k_ln<<<SEQ, 256, 0, stream>>>(x, ln1g, ln1b, hb);
  // QKV GEMM: [4096,1024] x [1024,3072] -> f32
  k_gemm<0><<<(SEQ / 32) * (H3 / 64) / 8, 256, 0, stream>>>(hb, Wqkv_s, qkv, nullptr, nullptr, nullptr, SEQ, H3, DM);
  // RoPE + V transpose
  k_rope<<<(NH * SEQ * 32) / 256, 256, 0, stream>>>(qkv, Qb, Kb, Vt);
  // causal flash attention
  k_attn<<<(NH * (SEQ / 16)) / 8, 256, 0, stream>>>(Qb, Kb, Vt, Ob);
  // Wo GEMM + residual -> x1
  k_gemm<1><<<(SEQ / 32) * (DM / 64) / 8, 256, 0, stream>>>(Ob, Wo_s, x1, nullptr, nullptr, x, SEQ, DM, DM);
  // LN2
  k_ln<<<SEQ, 256, 0, stream>>>(x1, ln2g, ln2b, hb);
  // FFN up: +b1, exact GELU -> bf16
  k_gemm<2><<<(SEQ / 32) * (HFF / 64) / 8, 256, 0, stream>>>(hb, W1_s, nullptr, f1, b1, nullptr, SEQ, HFF, DM);
  // FFN down: +b2 + residual -> output f32
  k_gemm<3><<<(SEQ / 32) * (DM / 64) / 8, 256, 0, stream>>>(f1, W2_s, out, nullptr, b2, x1, SEQ, DM, HFF);
}
